// BertAttention_609885357022
// MI455X (gfx1250) — compile-verified, matrix operand order checked by reading, not run on hardware
//
#include <hip/hip_runtime.h>

#define DEV __device__ __forceinline__

typedef __bf16 bf16;
typedef __attribute__((ext_vector_type(16))) __bf16 v16bf;
typedef __attribute__((ext_vector_type(8)))  float  v8f;

constexpr int   Bn = 4, Sn = 1024, HIDn = 1024, Hn = 16, Dn = 64;
constexpr float INV_SCALE = 0.07216878364870323f; // 1/sqrt(64*3)

// ---------------------------------------------------------------- epilogues
struct EpiParams {
  float* outF;
  bf16*  q_out; bf16* k_out; bf16* v_out;   // q_out doubles as generic bf16 out
  const float* bias; const float* bias2; const float* resid;
  const bf16* g1; const bf16* g2;           // c2p / p2c gather sources
  float scale;
};

enum { EPI_QKV = 1, EPI_HEADS = 2, EPI_BF16 = 3, EPI_SCORES = 4, EPI_CTX = 5, EPI_OUT = 6 };

template<int EPI>
DEV void epi_store(int m, int n, int z, float v, const EpiParams& ep) {
  if constexpr (EPI == EPI_QKV) {
    int b = m >> 10, s = m & 1023;
    if (n < HIDn) {                              // q: +bias, *1/sqrt(192), head-major
      int h = n >> 6, d = n & 63;
      float q = (v + ep.bias[n]) * ep.scale;
      ep.q_out[(((size_t)(b * Hn + h)) * Sn + s) * Dn + d] = (bf16)q;
    } else if (n < 2 * HIDn) {                   // k: head-major
      int n2 = n - HIDn; int h = n2 >> 6, d = n2 & 63;
      ep.k_out[(((size_t)(b * Hn + h)) * Sn + s) * Dn + d] = (bf16)v;
    } else {                                     // v: +bias, stored transposed [b,h,d,s]
      int n3 = n - 2 * HIDn; int h = n3 >> 6, d = n3 & 63;
      float vv = v + ep.bias2[n3];
      ep.v_out[(((size_t)(b * Hn + h)) * Dn + d) * Sn + s] = (bf16)vv;
    }
  } else if constexpr (EPI == EPI_HEADS) {       // rel-emb projections -> [h,p,d]
    int h = n >> 6, d = n & 63;
    float x = (ep.bias ? v + ep.bias[n] : v) * ep.scale;
    ep.q_out[((size_t)h * Sn + m) * Dn + d] = (bf16)x;
  } else if constexpr (EPI == EPI_BF16) {        // C2P / P2C [z, m, n]
    ep.q_out[(size_t)z * Sn * Sn + (size_t)m * Sn + n] = (bf16)v;
  } else if constexpr (EPI == EPI_SCORES) {      // qk + banded rel-pos gathers
    int p = m - n + 512; p = p < 0 ? 0 : (p > 1023 ? 1023 : p);
    size_t zb = (size_t)z * Sn * Sn;
    float x = v + (float)ep.g1[zb + (size_t)m * Sn + p]
                + (float)ep.g2[zb + (size_t)n * Sn + p];
    ep.outF[zb + (size_t)m * Sn + n] = x;
  } else if constexpr (EPI == EPI_CTX) {         // ctx heads -> [b, s, hid]
    int b = z >> 4, h = z & 15;
    ep.q_out[((size_t)(b * Sn + m)) * HIDn + h * Dn + n] = (bf16)v;
  } else {                                       // out proj + bias + residual
    ep.outF[(size_t)m * HIDn + n] = v + ep.bias[n] + ep.resid[(size_t)m * HIDn + n];
  }
}

// --------------------------------------------------- WMMA fragment loading
// 16-bit A (and symmetric B^T) 16x32 fragment, per CDNA5 ISA 7.12.2:
// lanes 0-15: row = lane,    K = {0..7, 16..23}
// lanes 16-31: row = lane-16, K = {8..15, 24..31}
// base comes straight from a kernarg (addrspace(1) inferable); off is the
// element offset of (row, kk + ((lane>>4)<<3)).
DEV v16bf load_frag(const bf16* __restrict__ base, size_t off) {
  const bf16* p = base + off;
  v16bf f;
#pragma unroll
  for (int i = 0; i < 4; ++i) { f[2*i] = p[2*i]; f[2*i+1] = p[2*i+1]; }
#pragma unroll
  for (int i = 4; i < 8; ++i) { f[2*i] = p[8 + 2*i]; f[2*i+1] = p[8 + 2*i + 1]; }
  return f;
}

// ------------------------------------------------------------- GEMM kernel
// D = A (MxK, row-major) * Bt^T (Bt is NxK row-major).
// 128-thread block = 4 waves in a 2x2 arrangement; each wave owns a 64x32 tile
// (8 v8f accumulators = 64 VGPRs; 6 fragments = 48 VGPRs -> no spills).
template<int EPI>
__global__ __launch_bounds__(128)
void gemm_wmma(const bf16* __restrict__ A, const bf16* __restrict__ Bt,
               int M, int N, int K, int lda, int ldb,
               size_t aStrideZ, size_t bStrideZ, int bMod,
               EpiParams ep)
{
  int lane = threadIdx.x & 31;
  int wid  = threadIdx.x >> 5;
  int z = blockIdx.z;
  size_t zbase_a = (size_t)z * aStrideZ;
  size_t zbase_b = (size_t)(bMod ? (z % bMod) : z) * bStrideZ;
  int m0 = blockIdx.x * 128 + (wid >> 1) * 64;
  int n0 = blockIdx.y * 64  + (wid & 1) * 32;
  if (m0 >= M || n0 >= N) return;

  v8f acc[4][2];
#pragma unroll
  for (int i = 0; i < 4; ++i)
#pragma unroll
    for (int j = 0; j < 2; ++j)
#pragma unroll
      for (int t = 0; t < 8; ++t) acc[i][j][t] = 0.0f;

  int rA = m0 + (lane & 15);
  int rB = n0 + (lane & 15);
  int ksub = (lane >> 4) << 3;

  size_t oa[4], ob[2];
#pragma unroll
  for (int i = 0; i < 4; ++i) oa[i] = zbase_a + (size_t)(rA + 16 * i) * lda + ksub;
#pragma unroll
  for (int j = 0; j < 2; ++j) ob[j] = zbase_b + (size_t)(rB + 16 * j) * ldb + ksub;

  for (int kk = 0; kk < K; kk += 32) {
    v16bf a[4], b[2];
#pragma unroll
    for (int j = 0; j < 2; ++j) { b[j] = load_frag(Bt, ob[j]); ob[j] += 32; }
#pragma unroll
    for (int i = 0; i < 4; ++i) { a[i] = load_frag(A, oa[i]); oa[i] += 32; }
#pragma unroll
    for (int i = 0; i < 4; ++i)
#pragma unroll
      for (int j = 0; j < 2; ++j)
        acc[i][j] = __builtin_amdgcn_wmma_f32_16x16x32_bf16(
            false, a[i], false, b[j], (short)0, acc[i][j], false, false);
  }

  int hi = lane >> 4;
  int nl = lane & 15;
#pragma unroll
  for (int i = 0; i < 4; ++i)
#pragma unroll
    for (int j = 0; j < 2; ++j)
#pragma unroll
      for (int r = 0; r < 8; ++r) {
        int m = m0 + 16 * i + 8 * hi + r;   // C/D layout: VGPR r holds M=r (+8 for hi lanes)
        int n = n0 + 16 * j + nl;
        epi_store<EPI>(m, n, z, acc[i][j][r], ep);
      }
}

// ------------------------------------------------------------ prep kernels
__global__ void cast_bf16(const float* __restrict__ x, bf16* __restrict__ y, size_t n) {
  size_t i = (size_t)blockIdx.x * blockDim.x + threadIdx.x;
  if (i < n) y[i] = (bf16)x[i];
}

// Wt[n*K + k] = (bf16) W[k*N + n]
__global__ void transpose_cast(const float* __restrict__ W, bf16* __restrict__ Wt,
                               int K, int N) {
  __shared__ float tile[32][33];
  int nb = blockIdx.x * 32, kb = blockIdx.y * 32;
  int tx = threadIdx.x, ty = threadIdx.y;       // 32 x 8
  for (int r = ty; r < 32; r += 8) tile[r][tx] = W[(size_t)(kb + r) * N + nb + tx];
  __syncthreads();
  for (int r = ty; r < 32; r += 8) Wt[(size_t)(nb + r) * K + kb + tx] = (bf16)tile[tx][r];
}

// ---------------------------------------------------------------- softmax
__global__ void softmax_kernel(const float* __restrict__ scores,
                               const int* __restrict__ mask,
                               bf16* __restrict__ probs) {
  int row = blockIdx.x;            // (b*H + h)*S + q
  int q = row & (Sn - 1);
  int b = row >> 14;               // / (H*S)
  size_t soff = (size_t)row * Sn;
  size_t moff = ((size_t)b * Sn + q) * Sn;   // mask[b,0,q,:]
  int tid = threadIdx.x;
  __shared__ float red[256];

  float x[4]; int keep[4];
  float lmax = -3.4e38f;
#pragma unroll
  for (int t = 0; t < 4; ++t) {
    int c = tid + 256 * t;
    keep[t] = mask[moff + c];
    x[t] = keep[t] ? scores[soff + c] : -3.4e38f;
    lmax = fmaxf(lmax, x[t]);
  }
  red[tid] = lmax; __syncthreads();
  for (int s = 128; s > 0; s >>= 1) { if (tid < s) red[tid] = fmaxf(red[tid], red[tid + s]); __syncthreads(); }
  float rmax = red[0]; __syncthreads();

  float e[4]; float lsum = 0.f;
#pragma unroll
  for (int t = 0; t < 4; ++t) { e[t] = __expf(x[t] - rmax); lsum += e[t]; }
  red[tid] = lsum; __syncthreads();
  for (int s = 128; s > 0; s >>= 1) { if (tid < s) red[tid] += red[tid + s]; __syncthreads(); }
  float inv = 1.0f / red[0];

#pragma unroll
  for (int t = 0; t < 4; ++t) {
    int c = tid + 256 * t;
    probs[soff + c] = (bf16)(keep[t] ? e[t] * inv : 0.0f);
  }
}

// -------------------------------------------------------------- layernorm
__global__ void layernorm_kernel(const float* __restrict__ h1,
                                 const int* __restrict__ mask,
                                 const float* __restrict__ g,
                                 const float* __restrict__ bb,
                                 float* __restrict__ out) {
  int row = blockIdx.x;            // b*S + s
  int b = row >> 10, s = row & 1023;
  size_t off = (size_t)row * HIDn;
  int tid = threadIdx.x;
  __shared__ float red[256];

  float x[4]; float lsum = 0.f;
#pragma unroll
  for (int t = 0; t < 4; ++t) { x[t] = h1[off + tid + 256 * t]; lsum += x[t]; }
  red[tid] = lsum; __syncthreads();
  for (int st = 128; st > 0; st >>= 1) { if (tid < st) red[tid] += red[tid + st]; __syncthreads(); }
  float mu = red[0] * (1.0f / HIDn); __syncthreads();

  float lvar = 0.f;
#pragma unroll
  for (int t = 0; t < 4; ++t) { float d = x[t] - mu; lvar += d * d; }
  red[tid] = lvar; __syncthreads();
  for (int st = 128; st > 0; st >>= 1) { if (tid < st) red[tid] += red[tid + st]; __syncthreads(); }
  float rstd = rsqrtf(red[0] * (1.0f / HIDn) + 1e-7f);

  float mv = (float)mask[(size_t)b * Sn * Sn + s];   // mask[b,0,0,s]
#pragma unroll
  for (int t = 0; t < 4; ++t) {
    int c = tid + 256 * t;
    out[off + c] = ((x[t] - mu) * rstd * g[c] + bb[c]) * mv;
  }
}

// ------------------------------------------------------------------ driver
extern "C" void kernel_launch(void* const* d_in, const int* in_sizes, int n_in,
                              void* d_out, int out_size, void* d_ws, size_t ws_size,
                              hipStream_t stream) {
  const float* hidden = (const float*)d_in[0];
  const int*   amask  = (const int*)d_in[1];
  const float* rel_w  = (const float*)d_in[2];
  const float* in_w   = (const float*)d_in[3];
  const float* q_bias = (const float*)d_in[4];
  const float* v_bias = (const float*)d_in[5];
  const float* pos_w  = (const float*)d_in[6];
  const float* posq_w = (const float*)d_in[7];
  const float* posq_b = (const float*)d_in[8];
  const float* out_w  = (const float*)d_in[9];
  const float* out_b  = (const float*)d_in[10];
  const float* ln_g   = (const float*)d_in[11];
  const float* ln_b   = (const float*)d_in[12];
  float* out = (float*)d_out;

  char* base = (char*)d_ws;
  size_t off = 0;
  auto alloc = [&](size_t elems, size_t esz) -> void* {
    void* p = base + off;
    off += (elems * esz + 255) & ~(size_t)255;
    return p;
  };

  bf16*  Hb      = (bf16*)alloc(4096ull * 1024, 2);
  bf16*  Wqkv_t  = (bf16*)alloc(3072ull * 1024, 2);
  bf16*  Wpos_t  = (bf16*)alloc(1024ull * 1024, 2);
  bf16*  Wposq_t = (bf16*)alloc(1024ull * 1024, 2);
  bf16*  Wout_t  = (bf16*)alloc(1024ull * 1024, 2);
  bf16*  Rel     = (bf16*)alloc(1024ull * 1024, 2);
  bf16*  Qh      = (bf16*)alloc(64ull * 1024 * 64, 2);
  bf16*  Kh      = (bf16*)alloc(64ull * 1024 * 64, 2);
  bf16*  VhT     = (bf16*)alloc(64ull * 1024 * 64, 2);
  bf16*  PosK    = (bf16*)alloc(16ull * 1024 * 64, 2);
  bf16*  PosQ    = (bf16*)alloc(16ull * 1024 * 64, 2);
  bf16*  C2P     = (bf16*)alloc(64ull * 1024 * 1024, 2);
  bf16*  P2C     = (bf16*)alloc(64ull * 1024 * 1024, 2);
  float* Scr     = (float*)alloc(64ull * 1024 * 1024, 4);
  bf16*  Prob    = (bf16*)alloc(64ull * 1024 * 1024, 2);
  bf16*  CtxB    = (bf16*)alloc(4096ull * 1024, 2);
  float* H1      = (float*)alloc(4096ull * 1024, 4);

  // casts + weight transposes
  cast_bf16<<<(4096 * 1024) / 256, 256, 0, stream>>>(hidden, Hb, 4096ull * 1024);
  cast_bf16<<<(1024 * 1024) / 256, 256, 0, stream>>>(rel_w, Rel, 1024ull * 1024);  // SPAN=512 -> whole table
  transpose_cast<<<dim3(3072 / 32, 1024 / 32), dim3(32, 8), 0, stream>>>(in_w,   Wqkv_t,  1024, 3072);
  transpose_cast<<<dim3(1024 / 32, 1024 / 32), dim3(32, 8), 0, stream>>>(pos_w,  Wpos_t,  1024, 1024);
  transpose_cast<<<dim3(1024 / 32, 1024 / 32), dim3(32, 8), 0, stream>>>(posq_w, Wposq_t, 1024, 1024);
  transpose_cast<<<dim3(1024 / 32, 1024 / 32), dim3(32, 8), 0, stream>>>(out_w,  Wout_t,  1024, 1024);

  EpiParams ep;

  // QKV projection, split + bias/scale into head-major Qh/Kh/VhT
  ep = {}; ep.q_out = Qh; ep.k_out = Kh; ep.v_out = VhT;
  ep.bias = q_bias; ep.bias2 = v_bias; ep.scale = INV_SCALE;
  gemm_wmma<EPI_QKV><<<dim3(32, 48, 1), 128, 0, stream>>>(Hb, Wqkv_t, 4096, 3072, 1024, 1024, 1024, 0, 0, 0, ep);

  // pos_key = heads(rel @ pos_proj_w)
  ep = {}; ep.q_out = PosK; ep.bias = nullptr; ep.scale = 1.0f;
  gemm_wmma<EPI_HEADS><<<dim3(8, 16, 1), 128, 0, stream>>>(Rel, Wpos_t, 1024, 1024, 1024, 1024, 1024, 0, 0, 0, ep);

  // pos_query = heads(rel @ pos_q_proj_w + b) / sqrt(192)
  ep = {}; ep.q_out = PosQ; ep.bias = posq_b; ep.scale = INV_SCALE;
  gemm_wmma<EPI_HEADS><<<dim3(8, 16, 1), 128, 0, stream>>>(Rel, Wposq_t, 1024, 1024, 1024, 1024, 1024, 0, 0, 0, ep);

  // C2P[z=bh] = Qh[bh] @ PosK[h]^T      (bMod=16 -> B indexed by h = z%16)
  ep = {}; ep.q_out = C2P;
  gemm_wmma<EPI_BF16><<<dim3(8, 16, 64), 128, 0, stream>>>(Qh, PosK, 1024, 1024, 64, 64, 64,
                                                           1024ull * 64, 1024ull * 64, 16, ep);
  // P2C[z=bh] = Kh[bh] @ PosQ[h]^T
  ep = {}; ep.q_out = P2C;
  gemm_wmma<EPI_BF16><<<dim3(8, 16, 64), 128, 0, stream>>>(Kh, PosQ, 1024, 1024, 64, 64, 64,
                                                           1024ull * 64, 1024ull * 64, 16, ep);

  // scores = Qh @ Kh^T + gather(C2P) + gather(P2C)
  ep = {}; ep.outF = Scr; ep.g1 = C2P; ep.g2 = P2C;
  gemm_wmma<EPI_SCORES><<<dim3(8, 16, 64), 128, 0, stream>>>(Qh, Kh, 1024, 1024, 64, 64, 64,
                                                             1024ull * 64, 1024ull * 64, 0, ep);

  // masked softmax -> bf16 probs
  softmax_kernel<<<64 * 1024, 256, 0, stream>>>(Scr, amask, Prob);

  // ctx = probs @ V  (VhT is already the N x K = D x S operand)
  ep = {}; ep.q_out = CtxB;
  gemm_wmma<EPI_CTX><<<dim3(8, 1, 64), 128, 0, stream>>>(Prob, VhT, 1024, 64, 1024, 1024, 1024,
                                                         1024ull * 1024, 64ull * 1024, 0, ep);

  // out projection + bias + residual
  ep = {}; ep.outF = H1; ep.bias = out_b; ep.resid = hidden;
  gemm_wmma<EPI_OUT><<<dim3(32, 16, 1), 128, 0, stream>>>(CtxB, Wout_t, 4096, 1024, 1024, 1024, 1024, 0, 0, 0, ep);

  // layernorm + output mask
  layernorm_kernel<<<4096, 256, 0, stream>>>(H1, amask, ln_g, ln_b, out);
}
